// MultiHeadAttention_87754771791884
// MI455X (gfx1250) — compile-verified, matrix operand order checked by reading, not run on hardware
//
#include <hip/hip_runtime.h>
#include <hip/hip_bf16.h>

// ---------------------------------------------------------------------------
// MHA forward for MI455X (gfx1250): f16 WMMA everywhere, f32 accumulate.
// Fixed problem size from the reference: B=4, S=2048, D=1024, H=16, d=64.
// ---------------------------------------------------------------------------

typedef __attribute__((ext_vector_type(16))) _Float16 v16h;
typedef __attribute__((ext_vector_type(8)))  _Float16 h8;
typedef __attribute__((ext_vector_type(4)))  _Float16 h4;
typedef __attribute__((ext_vector_type(8)))  float    v8f;

constexpr int B_  = 4;
constexpr int S_  = 2048;
constexpr int D_  = 1024;
constexpr int H_  = 16;
constexpr int HD_ = 64;          // per-head depth
constexpr int GM  = B_ * S_;     // 8192  (GEMM M)
constexpr int GN  = D_;          // 1024  (GEMM N)
constexpr int GK  = D_;          // 1024  (GEMM K)

// WMMA wrapper: D = A(16x32 f16) * B(32x16 f16) + C(f32)
__device__ __forceinline__ v8f wmma16(v16h a, v16h b, v8f c) {
  return __builtin_amdgcn_wmma_f32_16x16x32_f16(
      /*neg_a=*/false, a, /*neg_b=*/false, b,
      /*c_mod=*/(short)0, c, /*reuse_a=*/false, /*reuse_b=*/false);
}

// LDS byte offset of a generic pointer known to alias LDS (addrspacecast to
// AS3 strips the flat aperture; AS3 pointers are 32-bit LDS offsets).
__device__ __forceinline__ unsigned lds_off_u32(const void* p) {
  return (unsigned)(__SIZE_TYPE__)(__attribute__((address_space(3))) const void*)p;
}

// CDNA5 async copy: global -> LDS, 16B per lane, tracked by ASYNCcnt
// (GLOBAL_LOAD_ASYNC_TO_LDS_B128, ISA 15.18.3 op 98).
__device__ __forceinline__ void async_g2l_b128(unsigned lds_byte_off, const void* gptr) {
  asm volatile("global_load_async_to_lds_b128 %0, %1, off"
               :: "v"(lds_byte_off), "v"(gptr)
               : "memory");
}
__device__ __forceinline__ void wait_async_lds() {
  asm volatile("s_wait_asynccnt 0x0" ::: "memory");
}

// Load a 16x32 f16 A-fragment (or B-fragment from an N-major/K-contiguous
// matrix) per ISA 7.12.2: lanes 0-15 hold row r, k={0..7,16..23};
// lanes 16-31 hold row r, k={8..15,24..31}.  base points at (row0,k0),
// ld = row stride in elements.
__device__ __forceinline__ v16h load_frag_f16(const _Float16* base, int ld, int lane) {
  const int r  = lane & 15;
  const int kk = (lane & 16) ? 8 : 0;
  const _Float16* p = base + (size_t)r * ld + kk;
  union { v16h v; h8 h[2]; } u;
  u.h[0] = *(const h8*)(p);
  u.h[1] = *(const h8*)(p + 16);
  return u.v;
}

// Same fragment but sourced from f32 (LDS logits), converting to f16.
__device__ __forceinline__ v16h load_frag_f32cvt(const float* base, int ld, int lane) {
  const int r  = lane & 15;
  const int kk = (lane & 16) ? 8 : 0;
  const float* p = base + (size_t)r * ld + kk;
  v16h v;
#pragma unroll
  for (int i = 0; i < 8; ++i) {
    v[i]     = (_Float16)p[i];
    v[i + 8] = (_Float16)p[i + 16];
  }
  return v;
}

// ---------------------------------------------------------------------------
// Generic projection GEMM:  P = A(MxK) @ W(KxN) + bias
//   AT = float    -> convert to f16 while staging (VGPR path)
//   AT = _Float16 -> async global->LDS staging (ASYNCcnt path)
// OUT_MODE: 0 = f16 row-major [m*GN+n]        (Qp, Kp)
//           1 = f16 per-head transposed VpT [((b*H+h)*64+dd)*S + s]
//           2 = f32 row-major, non-temporal   (final out, write-once)
// Tile: 128x128 per WG, 8 waves in 2x4, each wave 64x32 (4x2 WMMA tiles),
// K staged 32 at a time.  LDS strides padded to 40 halves (bank-conflict free
// 16-lane b128 reads: 40*2B stride -> bank step 20, all 16 distinct mod 64).
// ---------------------------------------------------------------------------
template <typename AT, int OUT_MODE>
__global__ __launch_bounds__(256) void gemm_proj_kernel(
    const AT* __restrict__ A, const float* __restrict__ W,
    const float* __restrict__ bias, void* __restrict__ outp) {
  __shared__ _Float16 As[128 * 40];   // [m][k], k-contiguous
  __shared__ _Float16 Ws[128 * 40];   // [n][k], k-contiguous (transposed W)

  const int tid  = threadIdx.x;
  const int lane = tid & 31;
  const int wv   = tid >> 5;
  const int wm   = wv >> 2;          // 0..1 : 64-row slab
  const int wn   = wv & 3;           // 0..3 : 32-col slab
  const int m0   = blockIdx.y * 128;
  const int n0   = blockIdx.x * 128;

  v8f acc[4][2];
#pragma unroll
  for (int i = 0; i < 4; ++i)
#pragma unroll
    for (int j = 0; j < 2; ++j) acc[i][j] = v8f{0.f,0.f,0.f,0.f,0.f,0.f,0.f,0.f};

  for (int k0 = 0; k0 < GK; k0 += 32) {
    __syncthreads();
    // ---- stage A tile: 128 rows x 32 k ----
    if constexpr (sizeof(AT) == 4) {
#pragma unroll
      for (int i = 0; i < 4; ++i) {
        const int lin = tid + i * 256;        // 0..1023 float4 chunks
        const int row = lin >> 3;
        const int c4  = (lin & 7) << 2;
        const float4 a4 = *(const float4*)(&A[(size_t)(m0 + row) * GK + k0 + c4]);
        h4 hv = { (_Float16)a4.x, (_Float16)a4.y, (_Float16)a4.z, (_Float16)a4.w };
        *(h4*)(As + row * 40 + c4) = hv;
      }
    } else {
      // f16 A: direct async global->LDS copy, no VGPR round trip.
#pragma unroll
      for (int i = 0; i < 2; ++i) {
        const int lin = tid + i * 256;        // 0..511 half8 chunks
        const int row = lin >> 2;
        const int c8  = (lin & 3) << 3;
        async_g2l_b128(lds_off_u32(As + row * 40 + c8),
                       &A[(size_t)(m0 + row) * GK + k0 + c8]);
      }
    }
    // ---- stage W tile transposed: Ws[n][k], 32 k-rows x 128 n-cols ----
#pragma unroll
    for (int i = 0; i < 4; ++i) {
      const int lin  = tid + i * 256;         // 0..1023 float4 chunks
      const int krow = lin >> 5;
      const int c4   = (lin & 31) << 2;
      const float4 w4 = *(const float4*)(&W[(size_t)(k0 + krow) * GN + n0 + c4]);
      Ws[(c4 + 0) * 40 + krow] = (_Float16)w4.x;
      Ws[(c4 + 1) * 40 + krow] = (_Float16)w4.y;
      Ws[(c4 + 2) * 40 + krow] = (_Float16)w4.z;
      Ws[(c4 + 3) * 40 + krow] = (_Float16)w4.w;
    }
    if constexpr (sizeof(AT) == 2) wait_async_lds();
    __syncthreads();

    v16h af[4], bf[2];
#pragma unroll
    for (int mi = 0; mi < 4; ++mi)
      af[mi] = load_frag_f16(As + (wm * 64 + mi * 16) * 40, 40, lane);
#pragma unroll
    for (int ni = 0; ni < 2; ++ni)
      bf[ni] = load_frag_f16(Ws + (wn * 32 + ni * 16) * 40, 40, lane);
#pragma unroll
    for (int mi = 0; mi < 4; ++mi)
#pragma unroll
      for (int ni = 0; ni < 2; ++ni)
        acc[mi][ni] = wmma16(af[mi], bf[ni], acc[mi][ni]);
  }

  // ---- epilogue: bias + store per OUT_MODE ----
  const int n_l   = lane & 15;
  const int m_off = (lane & 16) ? 8 : 0;
#pragma unroll
  for (int mi = 0; mi < 4; ++mi) {
#pragma unroll
    for (int ni = 0; ni < 2; ++ni) {
      const int gm0 = m0 + wm * 64 + mi * 16 + m_off;
      const int gn  = n0 + wn * 32 + ni * 16 + n_l;
      const float bb = bias[gn];
#pragma unroll
      for (int i = 0; i < 8; ++i) {
        const int gm = gm0 + i;
        const float val = acc[mi][ni][i] + bb;
        if constexpr (OUT_MODE == 0) {
          ((_Float16*)outp)[(size_t)gm * GN + gn] = (_Float16)val;
        } else if constexpr (OUT_MODE == 1) {
          const int b  = gm >> 11;          // / S_
          const int s  = gm & (S_ - 1);
          const int h  = gn >> 6;           // / HD_
          const int dd = gn & (HD_ - 1);
          ((_Float16*)outp)[((size_t)(b * H_ + h) * HD_ + dd) * S_ + s] = (_Float16)val;
        } else {
          // write-once final output: stream past L2 (TH=NT)
          __builtin_nontemporal_store(val, &((float*)outp)[(size_t)gm * GN + gn]);
        }
      }
    }
  }
}

// ---------------------------------------------------------------------------
// Attention kernel: one WG = 32 query rows of one (b,h).
//  LDS: full 32x2048 f32 logits stripe (stride 2052 to kill bank conflicts)
//       + 32x64 f16 Q tile (stride 72) + 32 f32 inv-sums.  ~261 KB < 320 KB.
//  Phase 1: logits = (Q@K^T)/8 + mask        (WMMA, 8 waves x 16 key-blocks)
//  Phase 2a: per-row max / exp / sum (8 lanes per row, shfl reductions)
//  Phase 2b: coalesced normalize + SINGLE non-temporal write of attn (1 GB,
//            write-once -> keep it out of the 192MB L2 so K/V tiles stay hot)
//  Phase 3: ctx = attn @ Vp                  (WMMA, A from LDS f32->f16)
// ---------------------------------------------------------------------------
constexpr int LSTR = S_ + 4;   // 2052 f32: stride mod 64 banks = 4
constexpr int QSTR = 72;       // f16 Q stride

__global__ __launch_bounds__(256) void attn_mha_kernel(
    const _Float16* __restrict__ Qp, const _Float16* __restrict__ Kp,
    const _Float16* __restrict__ VpT, const int* __restrict__ mask,
    float* __restrict__ attn_out, _Float16* __restrict__ ctx) {
  extern __shared__ float smem[];
  float*     ls   = smem;                          // [32][LSTR]
  _Float16*  Qs   = (_Float16*)(smem + 32 * LSTR); // [32][QSTR]
  float*     invb = (float*)(Qs + 32 * QSTR);      // [32]

  const int tid  = threadIdx.x;
  const int lane = tid & 31;
  const int wv   = tid >> 5;
  const int q0   = blockIdx.x * 32;
  const int h    = blockIdx.y;
  const int b    = blockIdx.z;

  // ---- stage Q tile (32 x 64 f16) via async global->LDS ----
  {
    const int row = tid >> 3;
    const int c8  = (tid & 7) << 3;
    async_g2l_b128(lds_off_u32(Qs + row * QSTR + c8),
                   &Qp[(size_t)(b * S_ + q0 + row) * D_ + h * HD_ + c8]);
    wait_async_lds();
  }
  __syncthreads();

  // ---- phase 1: logits ----
  v16h aq[2][2];
#pragma unroll
  for (int m = 0; m < 2; ++m)
#pragma unroll
    for (int ks = 0; ks < 2; ++ks)
      aq[m][ks] = load_frag_f16(Qs + (m * 16) * QSTR + ks * 32, QSTR, lane);

  const _Float16* Kbase = Kp + (size_t)(b * S_) * D_ + h * HD_;
  const int nb0 = wv * 16;
  for (int nb = nb0; nb < nb0 + 16; ++nb) {
    const int n0 = nb * 16;
    if (nb + 1 < nb0 + 16)
      __builtin_prefetch(Kbase + (size_t)(n0 + 16) * D_, 0, 1);
    const v16h bk0 = load_frag_f16(Kbase + (size_t)n0 * D_ + 0,  D_, lane);
    const v16h bk1 = load_frag_f16(Kbase + (size_t)n0 * D_ + 32, D_, lane);
    v8f acc0 = v8f{0.f,0.f,0.f,0.f,0.f,0.f,0.f,0.f};
    v8f acc1 = v8f{0.f,0.f,0.f,0.f,0.f,0.f,0.f,0.f};
    acc0 = wmma16(aq[0][0], bk0, acc0);
    acc0 = wmma16(aq[0][1], bk1, acc0);
    acc1 = wmma16(aq[1][0], bk0, acc1);
    acc1 = wmma16(aq[1][1], bk1, acc1);

    const int n_l   = lane & 15;
    const int m_off = (lane & 16) ? 8 : 0;
    const int key   = n0 + n_l;
    const float mv  = (float)mask[b * S_ + key] * (-1e9f);
#pragma unroll
    for (int i = 0; i < 8; ++i) {
      ls[(m_off + i)      * LSTR + key] = acc0[i] * 0.125f + mv;  // 1/sqrt(64)
      ls[(16 + m_off + i) * LSTR + key] = acc1[i] * 0.125f + mv;
    }
  }
  __syncthreads();

  // ---- phase 2a: row max / exp / sum (8 lanes per row) ----
  {
    const int r   = lane >> 3;        // 0..3
    const int row = wv * 4 + r;       // 0..31
    const int sub = lane & 7;
    float* lrow = ls + (size_t)row * LSTR;

    float mx = -3.4e38f;
    for (int c = sub; c < S_; c += 8) mx = fmaxf(mx, lrow[c]);
#pragma unroll
    for (int off = 4; off > 0; off >>= 1) mx = fmaxf(mx, __shfl_xor(mx, off, 8));

    float sum = 0.f;
    for (int c = sub; c < S_; c += 8) {
      const float e = __expf(lrow[c] - mx);
      lrow[c] = e;
      sum += e;
    }
#pragma unroll
    for (int off = 4; off > 0; off >>= 1) sum += __shfl_xor(sum, off, 8);
    if (sub == 0) invb[row] = 1.f / sum;
  }
  __syncthreads();

  // ---- phase 2b: coalesced normalize + non-temporal attn emission ----
  {
    float* abase = attn_out + ((size_t)(b * H_ + h) * S_ + q0) * S_;
#pragma unroll 4
    for (int idx = tid; idx < 32 * S_; idx += 256) {
      const int row = idx >> 11;           // / S_
      const int col = idx & (S_ - 1);
      const float p = ls[row * LSTR + col] * invb[row];
      ls[row * LSTR + col] = p;
      __builtin_nontemporal_store(p, abase + (size_t)row * S_ + col);
    }
  }
  __syncthreads();

  // ---- phase 3: ctx = attn @ Vp ; wave -> one 16x16 ctx tile ----
  {
    const int mblk = wv >> 2;         // 0..1
    const int nblk = wv & 3;          // 0..3 (16 depth cols each)
    v8f acc = v8f{0.f,0.f,0.f,0.f,0.f,0.f,0.f,0.f};
    const _Float16* Vb = VpT + ((size_t)(b * H_ + h) * HD_ + nblk * 16) * S_;
    for (int k0 = 0; k0 < S_; k0 += 32) {
      const v16h a  = load_frag_f32cvt(ls + (size_t)(mblk * 16) * LSTR + k0, LSTR, lane);
      const v16h bv = load_frag_f16(Vb + k0, S_, lane);
      acc = wmma16(a, bv, acc);
    }
    const int n_l   = lane & 15;
    const int m_off = (lane & 16) ? 8 : 0;
    const int dd    = nblk * 16 + n_l;
#pragma unroll
    for (int i = 0; i < 8; ++i) {
      const int qrow = q0 + mblk * 16 + m_off + i;
      ctx[(size_t)(b * S_ + qrow) * D_ + h * HD_ + dd] = (_Float16)acc[i];
    }
  }
}

// ---------------------------------------------------------------------------
// Host launcher
// ---------------------------------------------------------------------------
extern "C" void kernel_launch(void* const* d_in, const int* in_sizes, int n_in,
                              void* d_out, int out_size, void* d_ws, size_t ws_size,
                              hipStream_t stream) {
  const float* q    = (const float*)d_in[0];
  const float* k    = (const float*)d_in[1];
  const float* v    = (const float*)d_in[2];
  const int*   mask = (const int*)  d_in[3];
  const float* wq   = (const float*)d_in[4];
  const float* bq   = (const float*)d_in[5];
  const float* wk   = (const float*)d_in[6];
  const float* bk   = (const float*)d_in[7];
  const float* wv_  = (const float*)d_in[8];
  const float* bv   = (const float*)d_in[9];
  const float* wo   = (const float*)d_in[10];
  const float* bo   = (const float*)d_in[11];

  float* out  = (float*)d_out;
  float* attn = out + (size_t)B_ * S_ * D_;          // tuple output #2

  const size_t projN = (size_t)GM * GN;              // 8.39M f16 elements
  _Float16* Qp  = (_Float16*)d_ws;
  _Float16* Kp  = Qp  + projN;
  _Float16* VpT = Kp  + projN;
  _Float16* ctx = VpT + projN;                       // 4 * 16 MB = 64 MB total

  const dim3 gGemm(GN / 128, GM / 128);              // (8, 64)
  const dim3 blk(256);

  // QKV projections (f32 in, f16 out; V stored transposed per head)
  gemm_proj_kernel<float, 0><<<gGemm, blk, 0, stream>>>(q, wq, bq, (void*)Qp);
  gemm_proj_kernel<float, 0><<<gGemm, blk, 0, stream>>>(k, wk, bk, (void*)Kp);
  gemm_proj_kernel<float, 1><<<gGemm, blk, 0, stream>>>(v, wv_, bv, (void*)VpT);

  // Fused attention (needs ~261 KB dynamic LDS; WGP has 320 KB)
  const size_t ldsBytes = (size_t)32 * LSTR * 4 + (size_t)32 * QSTR * 2 + 32 * 4;
  (void)hipFuncSetAttribute((const void*)attn_mha_kernel,
                            hipFuncAttributeMaxDynamicSharedMemorySize,
                            (int)ldsBytes);
  attn_mha_kernel<<<dim3(S_ / 32, H_, B_), blk, ldsBytes, stream>>>(
      Qp, Kp, VpT, mask, attn, ctx);

  // Output projection (f16 ctx in, f32 out, non-temporal)
  gemm_proj_kernel<_Float16, 2><<<gGemm, blk, 0, stream>>>(ctx, wo, bo, (void*)out);
}
